// GraphTransformerPlus_32049045962817
// MI455X (gfx1250) — matrix operand order repaired and checked by reading, not compiled, not run on hardware
//
#include <hip/hip_runtime.h>
#include <hip/hip_bf16.h>
#include <math.h>

typedef __attribute__((ext_vector_type(16))) _Float16 v16h;
typedef __attribute__((ext_vector_type(8)))  float    v8f;

// ---------------------------------------------------------------------------
// Model constants
// ---------------------------------------------------------------------------
#define T_LEN 240
#define BATCH 2
#define E_DIM 128
#define H_NUM 8
#define HD_DIM 16
#define FF_DIM 512
#define D_DIM 256
#define SEQ 80
#define SD_DIM 17
#define BH 16          // BATCH * H_NUM

// ---------------------------------------------------------------------------
// Async global->LDS (gfx1250). Inline asm sidesteps the clang builtin's
// cuda_device/cuda_shared address-space pointer sugar (which HIP source
// cannot spell). VDST = per-lane LDS byte address (low 32 bits of a generic
// pointer into the shared aperture, per ISA 10.2), VADDR = 64-bit global
// address. Completion is tracked by ASYNCcnt.
// ---------------------------------------------------------------------------
static __device__ __forceinline__ void async_lds_b128(const void* gsrc, void* ldst) {
#if defined(__gfx1250__)
  const unsigned lds_addr = (unsigned)(unsigned long long)ldst;
  const unsigned long long gaddr = (unsigned long long)gsrc;
  asm volatile("global_load_async_to_lds_b128 %0, %1, off"
               :: "v"(lds_addr), "v"(gaddr) : "memory");
#else
  *(float4*)ldst = *(const float4*)gsrc;
#endif
}

static __device__ __forceinline__ void wait_async_zero() {
#if defined(__gfx1250__)
  asm volatile("s_wait_asynccnt 0x0" ::: "memory");
#endif
}

// ---------------------------------------------------------------------------
// Generic WMMA GEMM:  C[M,N] = epilogue(A[M,K] @ B + bias) (+ residual)
//   opB = 0 : B is K x N row-major      (Conv1D style  x @ W)
//   opB = 1 : B is N x K row-major      (F.linear style x @ W.T)
//   act = 0 none, 1 relu, 2 exact gelu
// One wave (32 threads) per 16x16 output tile. K multiple of 32,
// M and N multiples of 16 (true for every call site here).
// ---------------------------------------------------------------------------
__global__ void gemm_wmma(const float* __restrict__ A, const float* __restrict__ B,
                          const float* __restrict__ bias, const float* __restrict__ R,
                          float* __restrict__ C, int M, int N, int K, int opB, int act) {
  const int lane = threadIdx.x & 31;
  const int n0   = blockIdx.x * 16;
  const int row0 = blockIdx.y * 16;
  const int m    = lane & 15;
  const int koff = (lane >> 4) * 8;     // A: lanes 16-31 start at K+8
  const int nn   = n0 + (lane & 15);    // B/C: lane%16 = N column

  const float* arow = A + (size_t)(row0 + m) * K;
  v8f acc = {};

  for (int k0 = 0; k0 < K; k0 += 32) {
    if (k0 + 96 < K) __builtin_prefetch(arow + k0 + 96, 0, 1);
    v16h av, bv;
#pragma unroll
    for (int h = 0; h < 8; ++h)  av[h]     = (_Float16)arow[k0 + koff + h];
#pragma unroll
    for (int h = 0; h < 8; ++h)  av[8 + h] = (_Float16)arow[k0 + koff + 16 + h];
    const int kk = k0 + (lane >> 4) * 16;
    if (opB == 0) {
#pragma unroll
      for (int h = 0; h < 16; ++h) bv[h] = (_Float16)B[(size_t)(k0 + koff + h + (h & 8)) * N + nn];
    } else {
#pragma unroll
      for (int h = 0; h < 16; ++h) bv[h] = (_Float16)B[(size_t)nn * K + k0 + koff + h + (h & 8)];
    }
    acc = __builtin_amdgcn_wmma_f32_16x16x32_f16(false, av, false, bv,
                                                 (short)0, acc, false, false);
  }

#pragma unroll
  for (int jj = 0; jj < 8; ++jj) {
    const int r = row0 + jj + (lane >> 4) * 8;
    float v = acc[jj];
    if (bias) v += bias[nn];
    if (act == 1)      v = fmaxf(v, 0.f);
    else if (act == 2) v = 0.5f * v * (1.f + erff(v * 0.70710678118654752f));
    if (R) v += R[(size_t)r * N + nn];
    C[(size_t)r * N + nn] = v;
  }
}

// ---------------------------------------------------------------------------
// Fused relation-projection + attention-logit kernel.
// Block (j, i-tile of 16): DMA-stage rel[j, i0..i0+15, b=0/1, :] (32x128 f32,
// 16 KB) into LDS with async global->LDS b128 transfers, project through
// rel_w (128x256) with WMMA (A fragments read from LDS), keep the 32x256
// projection in LDS, then emit
//   logits[i, j, bh] = 0.0625 * sum_n (q[i,bh,n]+RA)(k[j,bh,n]+RB)
// (reference applies 1/sqrt(HD) twice -> 0.0625). The 118 MB "r" tensor is
// never materialized in HBM; rel is streamed from HBM exactly once per block.
// ---------------------------------------------------------------------------
__global__ void rel_logits_fused(const float* __restrict__ rel,   // (240,240,2,128)
                                 const float* __restrict__ rw,    // (128,256)
                                 const float* __restrict__ rb,    // (256)
                                 const float* __restrict__ qkv,   // (240,2,384)
                                 float* __restrict__ logits) {    // (240*16,240)
  __shared__ float a_s[32][E_DIM];   // staged rel tile   (16 KB)
  __shared__ float r_s[32][D_DIM];   // projection result (32 KB)

  const int j    = blockIdx.x;        // source token
  const int i0   = blockIdx.y * 16;   // dest-token tile base
  const int t    = threadIdx.x;
  const int wave = t >> 5;
  const int lane = t & 31;
  const int mt   = wave & 1;          // batch handled by this wave (M-tile)
  const int ngrp = wave >> 1;         // 4 column groups of 4 N-tiles

  // ---- stage A tile: 1024 x 16B chunks, 4 per thread ---------------------
  // LDS row rr = b*16 + i_local  <->  rel[j, i0+i_local, b, :]
#pragma unroll
  for (int p = 0; p < 4; ++p) {
    const int idx = t + p * 256;
    const int rr  = idx >> 5;          // 0..31
    const int c   = idx & 31;          // 16B chunk within the 128-float row
    const float* src = rel +
        (((size_t)j * T_LEN + (i0 + (rr & 15))) * BATCH + (rr >> 4)) * E_DIM + c * 4;
    async_lds_b128(src, &a_s[rr][c * 4]);
  }
  wait_async_zero();
  __syncthreads();

  // ---- WMMA phase: 16 WMMAs per wave, A fragments from LDS ---------------
  const int m    = lane & 15;
  const int koff = (lane >> 4) * 8;
  const float* arow = &a_s[mt * 16 + m][0];

  v8f acc[4] = {{}, {}, {}, {}};
#pragma unroll
  for (int ks = 0; ks < 4; ++ks) {
    const int k0 = ks * 32;
    v16h av;
#pragma unroll
    for (int h = 0; h < 8; ++h)  av[h]     = (_Float16)arow[k0 + koff + h];
#pragma unroll
    for (int h = 0; h < 8; ++h)  av[8 + h] = (_Float16)arow[k0 + koff + 16 + h];
    const int kk = k0 + (lane >> 4) * 16;
#pragma unroll
    for (int tt = 0; tt < 4; ++tt) {
      const int nn = (ngrp * 4 + tt) * 16 + (lane & 15);
      v16h bv;
#pragma unroll
      for (int h = 0; h < 16; ++h) bv[h] = (_Float16)rw[(size_t)(k0 + koff + h + (h & 8)) * D_DIM + nn];
      acc[tt] = __builtin_amdgcn_wmma_f32_16x16x32_f16(false, av, false, bv,
                                                       (short)0, acc[tt], false, false);
    }
  }
  __syncthreads();   // a_s reads complete before r_s phase
  // stash projection (+bias) into LDS: row = b*16 + i_local
#pragma unroll
  for (int tt = 0; tt < 4; ++tt) {
    const int col = (ngrp * 4 + tt) * 16 + (lane & 15);
#pragma unroll
    for (int jj = 0; jj < 8; ++jj) {
      const int rowm = jj + (lane >> 4) * 8;
      r_s[mt * 16 + rowm][col] = acc[tt][jj] + rb[col];
    }
  }
  __syncthreads();

  // ---- combine phase: 256 threads -> 16 i x 16 bh logits for this j ------
  const int il = t >> 4;
  const int bh = t & 15;
  const int bb = bh >> 3;
  const int hh = bh & 7;
  const int i  = i0 + il;
  const int row = bb * 16 + il;
  const float* qrow = qkv + ((size_t)i * BATCH + bb) * 384 + hh * HD_DIM;          // q
  const float* krow = qkv + ((size_t)j * BATCH + bb) * 384 + E_DIM + hh * HD_DIM;  // k
  float s = 0.f;
#pragma unroll
  for (int n = 0; n < HD_DIM; ++n) {
    const float qa = qrow[n] + r_s[row][hh * HD_DIM + n];
    const float kb = krow[n] + r_s[row][E_DIM + hh * HD_DIM + n];
    s += qa * kb;
  }
  logits[((size_t)i * BH + bh) * T_LEN + j] = s * 0.0625f;
}

// ---------------------------------------------------------------------------
// In-place softmax over rows of length n (n <= 256, blockDim = 256)
// ---------------------------------------------------------------------------
__global__ void softmax_rows(float* __restrict__ w, int n) {
  __shared__ float red[256];
  float* row = w + (size_t)blockIdx.x * n;
  const int t = threadIdx.x;
  const float v = (t < n) ? row[t] : -INFINITY;
  red[t] = v; __syncthreads();
  for (int s = 128; s > 0; s >>= 1) { if (t < s) red[t] = fmaxf(red[t], red[t + s]); __syncthreads(); }
  const float mx = red[0]; __syncthreads();
  const float e = (t < n) ? expf(v - mx) : 0.f;
  red[t] = e; __syncthreads();
  for (int s = 128; s > 0; s >>= 1) { if (t < s) red[t] += red[t + s]; __syncthreads(); }
  const float sum = red[0];
  if (t < n) row[t] = e / sum;
}

// ---------------------------------------------------------------------------
// a[i, bh, n] = sum_j w[i,j,bh] * v[j,bh,n]  -> ATT (240,2,128)
// One block per i; 16 softmax rows staged in LDS.
// ---------------------------------------------------------------------------
__global__ void attn_out(const float* __restrict__ w, const float* __restrict__ qkv,
                         float* __restrict__ att) {
  __shared__ float ws[BH * T_LEN];
  const int i = blockIdx.x;
  const int t = threadIdx.x;
  for (int idx = t; idx < BH * T_LEN; idx += 256)
    ws[idx] = w[((size_t)i * BH + idx / T_LEN) * T_LEN + idx % T_LEN];
  __syncthreads();
  const int bh = t >> 4, n = t & 15;
  const int b = bh >> 3, h = bh & 7;
  const float* wr = ws + bh * T_LEN;
  float acc = 0.f;
  for (int jj = 0; jj < T_LEN; ++jj)
    acc += wr[jj] * qkv[((size_t)jj * BATCH + b) * 384 + 2 * E_DIM + h * HD_DIM + n];
  att[((size_t)i * BATCH + b) * E_DIM + h * HD_DIM + n] = acc;
}

// ---------------------------------------------------------------------------
// out[row] = LayerNorm(a[row] (+ b[row])) * g + beta ; blockDim == C (pow2)
// ---------------------------------------------------------------------------
__global__ void add_ln(const float* __restrict__ a, const float* __restrict__ b,
                       const float* __restrict__ g, const float* __restrict__ be,
                       float* __restrict__ out, int C) {
  extern __shared__ float red[];
  const int t = threadIdx.x;
  const size_t base = (size_t)blockIdx.x * C;
  float v = a[base + t] + (b ? b[base + t] : 0.f);
  red[t] = v; __syncthreads();
  for (int s = C >> 1; s > 0; s >>= 1) { if (t < s) red[t] += red[t + s]; __syncthreads(); }
  const float mean = red[0] / C; __syncthreads();
  const float d = v - mean;
  red[t] = d * d; __syncthreads();
  for (int s = C >> 1; s > 0; s >>= 1) { if (t < s) red[t] += red[t + s]; __syncthreads(); }
  const float var = red[0] / C;
  out[base + t] = d * rsqrtf(var + 1e-5f) * g[t] + be[t];
}

// g0 = state @ es_w + es_b + temb[:, :80]
__global__ void embed_global(const float* __restrict__ st, const float* __restrict__ ew,
                             const float* __restrict__ eb, const float* __restrict__ temb,
                             float* __restrict__ G) {
  const int row = blockIdx.x;        // b*80 + t
  const int n = threadIdx.x;         // 0..255
  const int tt = row % SEQ;
  float acc = eb[n] + temb[(size_t)tt * D_DIM + n];
  const float* sp = st + (size_t)row * SD_DIM;
  for (int s = 0; s < SD_DIM; ++s) acc += sp[s] * ew[(size_t)s * D_DIM + n];
  G[(size_t)row * D_DIM + n] = acc;
}

// local = x[::3].swapaxes(0,1) -> (2,80,128)
__global__ void extract_local(const float* __restrict__ X, float* __restrict__ L) {
  const int row = blockIdx.x;   // 0..159
  const int c = threadIdx.x;    // 0..127
  const int b = row / SEQ, tt = row % SEQ;
  L[(size_t)row * E_DIM + c] = X[((size_t)(3 * tt) * BATCH + b) * E_DIM + c];
}

// Causal global attention per (b, h, i): N=80, GHD=32
__global__ void global_attn(const float* __restrict__ Q, const float* __restrict__ Kk,
                            const float* __restrict__ V, float* __restrict__ O) {
  __shared__ float s[SEQ];
  __shared__ float shv[2];
  const int bid = blockIdx.x;
  const int i = bid % SEQ;
  const int h = (bid / SEQ) % H_NUM;
  const int b = bid / (SEQ * H_NUM);
  const int t = threadIdx.x;
  const float* qp = Q + ((size_t)b * SEQ + i) * D_DIM + h * 32;
  if (t < SEQ) {
    if (t <= i) {
      const float* kp = Kk + ((size_t)b * SEQ + t) * D_DIM + h * 32;
      float d = 0.f;
      for (int c = 0; c < 32; ++c) d += qp[c] * kp[c];
      s[t] = d * 0.17677669529663689f;   // 32^-0.5
    } else s[t] = -INFINITY;
  }
  __syncthreads();
  if (t == 0) { float mx = s[0]; for (int j = 1; j < SEQ; ++j) mx = fmaxf(mx, s[j]); shv[0] = mx; }
  __syncthreads();
  if (t < SEQ) s[t] = expf(s[t] - shv[0]);
  __syncthreads();
  if (t == 0) { float sm = 0.f; for (int j = 0; j < SEQ; ++j) sm += s[j]; shv[1] = sm; }
  __syncthreads();
  if (t < 32) {
    float acc = 0.f;
    for (int j = 0; j <= i; ++j)
      acc += s[j] * V[((size_t)b * SEQ + j) * D_DIM + h * 32 + t];
    O[((size_t)b * SEQ + i) * D_DIM + h * 32 + t] = acc / shv[1];
  }
}

// ---------------------------------------------------------------------------
// Host orchestration
// ---------------------------------------------------------------------------
extern "C" void kernel_launch(void* const* d_in, const int* in_sizes, int n_in,
                              void* d_out, int out_size, void* d_ws, size_t ws_size,
                              hipStream_t stream) {
  const float* x     = (const float*)d_in[0];
  const float* rel   = (const float*)d_in[1];
  const float* state = (const float*)d_in[2];
  const float* in_w  = (const float*)d_in[3];
  const float* in_b  = (const float*)d_in[4];
  const float* rel_w = (const float*)d_in[5];
  const float* rel_b = (const float*)d_in[6];
  const float* out_w = (const float*)d_in[7];
  const float* out_b = (const float*)d_in[8];
  const float* fc1_w = (const float*)d_in[9];
  const float* fc1_b = (const float*)d_in[10];
  const float* fc2_w = (const float*)d_in[11];
  const float* fc2_b = (const float*)d_in[12];
  const float* aln_g = (const float*)d_in[13];
  const float* aln_b = (const float*)d_in[14];
  const float* fln_g = (const float*)d_in[15];
  const float* fln_b = (const float*)d_in[16];
  const float* ln1_g = (const float*)d_in[17];
  const float* ln1_b = (const float*)d_in[18];
  const float* ln2_g = (const float*)d_in[19];
  const float* ln2_b = (const float*)d_in[20];
  const float* q_w   = (const float*)d_in[21];
  const float* q_b   = (const float*)d_in[22];
  const float* k_w   = (const float*)d_in[23];
  const float* k_b   = (const float*)d_in[24];
  const float* v_w   = (const float*)d_in[25];
  const float* v_b   = (const float*)d_in[26];
  const float* p_w   = (const float*)d_in[27];
  const float* p_b   = (const float*)d_in[28];
  const float* m1_w  = (const float*)d_in[29];
  const float* m1_b  = (const float*)d_in[30];
  const float* m2_w  = (const float*)d_in[31];
  const float* m2_b  = (const float*)d_in[32];
  const float* lgp_w = (const float*)d_in[33];
  const float* lgp_b = (const float*)d_in[34];
  const float* es_w  = (const float*)d_in[35];
  const float* es_b  = (const float*)d_in[36];
  const float* temb  = (const float*)d_in[37];

  // workspace carve-up (all fp32)
  float* p = (float*)d_ws;
  float* XBUF   = p; p += 240 * 2 * 128;   // 61440
  float* QKV    = p; p += 480 * 384;       // 184320
  float* LOGITS = p; p += 3840 * 240;      // 921600
  float* ATT    = p; p += 240 * 2 * 128;
  float* OPROJ  = p; p += 240 * 2 * 128;
  float* H1     = p; p += 480 * 512;
  float* H2     = p; p += 240 * 2 * 128;
  float* LOC    = p; p += 160 * 128;
  float* G      = p; p += 160 * 256;
  float* Y      = p; p += 160 * 256;
  float* QG     = p; p += 160 * 256;
  float* KG     = p; p += 160 * 256;
  float* VG     = p; p += 160 * 256;
  float* OG     = p; p += 160 * 256;
  float* Z      = p; p += 160 * 256;
  float* HG     = p; p += 160 * 1024;

  (void)hipMemcpyAsync(XBUF, x, (size_t)61440 * sizeof(float),
                       hipMemcpyDeviceToDevice, stream);
  embed_global<<<160, 256, 0, stream>>>(state, es_w, es_b, temb, G);

  for (int l = 0; l < 2; ++l) {
    const float* in_w_l  = in_w  + (size_t)l * 384 * 128;
    const float* in_b_l  = in_b  + (size_t)l * 384;
    const float* rel_w_l = rel_w + (size_t)l * 128 * 256;
    const float* rel_b_l = rel_b + (size_t)l * 256;
    const float* out_w_l = out_w + (size_t)l * 128 * 128;
    const float* out_b_l = out_b + (size_t)l * 128;
    const float* fc1_w_l = fc1_w + (size_t)l * 128 * 512;
    const float* fc1_b_l = fc1_b + (size_t)l * 512;
    const float* fc2_w_l = fc2_w + (size_t)l * 512 * 128;
    const float* fc2_b_l = fc2_b + (size_t)l * 128;
    const float* aln_g_l = aln_g + (size_t)l * 128;
    const float* aln_b_l = aln_b + (size_t)l * 128;
    const float* fln_g_l = fln_g + (size_t)l * 128;
    const float* fln_b_l = fln_b + (size_t)l * 128;
    const float* ln1_g_l = ln1_g + (size_t)l * 256;
    const float* ln1_b_l = ln1_b + (size_t)l * 256;
    const float* ln2_g_l = ln2_g + (size_t)l * 256;
    const float* ln2_b_l = ln2_b + (size_t)l * 256;
    const float* q_w_l   = q_w   + (size_t)l * 256 * 256;
    const float* q_b_l   = q_b   + (size_t)l * 256;
    const float* k_w_l   = k_w   + (size_t)l * 256 * 256;
    const float* k_b_l   = k_b   + (size_t)l * 256;
    const float* v_w_l   = v_w   + (size_t)l * 256 * 256;
    const float* v_b_l   = v_b   + (size_t)l * 256;
    const float* p_w_l   = p_w   + (size_t)l * 256 * 256;
    const float* p_b_l   = p_b   + (size_t)l * 256;
    const float* m1_w_l  = m1_w  + (size_t)l * 256 * 1024;
    const float* m1_b_l  = m1_b  + (size_t)l * 1024;
    const float* m2_w_l  = m2_w  + (size_t)l * 1024 * 256;
    const float* m2_b_l  = m2_b  + (size_t)l * 256;

    // ---- local (relation) transformer layer ----
    gemm_wmma<<<dim3(24, 30), 32, 0, stream>>>(XBUF, in_w_l, in_b_l, nullptr, QKV,
                                               480, 384, 128, 1, 0);
    rel_logits_fused<<<dim3(240, 15), 256, 0, stream>>>(rel, rel_w_l, rel_b_l, QKV, LOGITS);
    softmax_rows<<<3840, 256, 0, stream>>>(LOGITS, 240);
    attn_out<<<240, 256, 0, stream>>>(LOGITS, QKV, ATT);
    gemm_wmma<<<dim3(8, 30), 32, 0, stream>>>(ATT, out_w_l, out_b_l, nullptr, OPROJ,
                                              480, 128, 128, 0, 0);
    add_ln<<<480, 128, 128 * sizeof(float), stream>>>(XBUF, OPROJ, aln_g_l, aln_b_l, XBUF, 128);
    gemm_wmma<<<dim3(32, 30), 32, 0, stream>>>(XBUF, fc1_w_l, fc1_b_l, nullptr, H1,
                                               480, 512, 128, 0, 1);
    gemm_wmma<<<dim3(8, 30), 32, 0, stream>>>(H1, fc2_w_l, fc2_b_l, nullptr, H2,
                                              480, 128, 512, 0, 0);
    add_ln<<<480, 128, 128 * sizeof(float), stream>>>(XBUF, H2, fln_g_l, fln_b_l, XBUF, 128);

    // ---- global stream block ----
    extract_local<<<160, 128, 0, stream>>>(XBUF, LOC);
    gemm_wmma<<<dim3(16, 10), 32, 0, stream>>>(LOC, lgp_w, lgp_b, G, G,
                                               160, 256, 128, 0, 0);
    add_ln<<<160, 256, 256 * sizeof(float), stream>>>(G, nullptr, ln1_g_l, ln1_b_l, Y, 256);
    gemm_wmma<<<dim3(16, 10), 32, 0, stream>>>(Y, q_w_l, q_b_l, nullptr, QG, 160, 256, 256, 0, 0);
    gemm_wmma<<<dim3(16, 10), 32, 0, stream>>>(Y, k_w_l, k_b_l, nullptr, KG, 160, 256, 256, 0, 0);
    gemm_wmma<<<dim3(16, 10), 32, 0, stream>>>(Y, v_w_l, v_b_l, nullptr, VG, 160, 256, 256, 0, 0);
    global_attn<<<2 * 8 * 80, 128, 0, stream>>>(QG, KG, VG, OG);
    gemm_wmma<<<dim3(16, 10), 32, 0, stream>>>(OG, p_w_l, p_b_l, G, G, 160, 256, 256, 0, 0);
    add_ln<<<160, 256, 256 * sizeof(float), stream>>>(G, nullptr, ln2_g_l, ln2_b_l, Z, 256);
    gemm_wmma<<<dim3(64, 10), 32, 0, stream>>>(Z, m1_w_l, m1_b_l, nullptr, HG,
                                               160, 1024, 256, 0, 2);
    gemm_wmma<<<dim3(16, 10), 32, 0, stream>>>(HG, m2_w_l, m2_b_l, G, G,
                                               160, 256, 1024, 0, 0);
  }

  (void)hipMemcpyAsync(d_out, G, (size_t)160 * 256 * sizeof(float),
                       hipMemcpyDeviceToDevice, stream);
}
